// TSMultiHeadAttention_38886633898069
// MI455X (gfx1250) — matrix operand order repaired and checked by reading, not compiled, not run on hardware
//
#include <hip/hip_runtime.h>
#include <hip/hip_bf16.h>

// ---------------------------------------------------------------------------
// MI455X (gfx1250) fused multi-head attention with relative position bias.
//   scores = (Q Wq)(K Wk + R)^T / sqrt(D) + qr_bias/sqrt(D)   (Q pre-scaled)
// Flash-attention online softmax; GEMMs via v_wmma_f32_16x16x32_f16; K'/V
// tiles staged into LDS with double-buffered GLOBAL_LOAD_ASYNC_TO_LDS_B128
// (ASYNCcnt) when the toolchain exposes the builtin.
// ---------------------------------------------------------------------------

typedef _Float16 h8v  __attribute__((ext_vector_type(8)));
typedef _Float16 h16v __attribute__((ext_vector_type(16)));
typedef float    f8v  __attribute__((ext_vector_type(8)));
typedef int      v4i  __attribute__((ext_vector_type(4)));

#define WMMA_F16(a, b, c) \
    __builtin_amdgcn_wmma_f32_16x16x32_f16(false, (a), false, (b), (short)0, (c), false, false)

#if __has_builtin(__builtin_amdgcn_global_load_async_to_lds_b128) && \
    __has_builtin(__builtin_amdgcn_s_wait_asynccnt)
#define USE_ASYNC_LDS 1
#else
#define USE_ASYNC_LDS 0
#endif

#define AS1 __attribute__((address_space(1)))
#define AS3 __attribute__((address_space(3)))

namespace cfg {
constexpr int B = 4, T = 2048, F = 512, H = 8, D = 64;
constexpr int HD = H * D;        // 512
constexpr int BT = B * T;        // 8192
}

// --- fragment helpers (16-bit WMMA operands in contiguous k order, same for A and B) ---
// A 16x32: lane l -> row m=l&15; kbase=(l<16)?0:16; elems 0..15 = K kbase..kbase+15
//          (contiguous).
__device__ __forceinline__ h16v frag_a_f16(const _Float16* row, int kb) {
    h8v lo = *(const h8v*)(row + kb);
    h8v hi = *(const h8v*)(row + kb + 8);
    return __builtin_shufflevector(lo, hi, 0, 1, 2, 3, 4, 5, 6, 7, 8, 9, 10, 11, 12, 13, 14, 15);
}
__device__ __forceinline__ h16v frag_a_f32(const float* row, int kb) {
    f8v lo = *(const f8v*)(row + kb);
    f8v hi = *(const f8v*)(row + kb + 8);
    h8v clo = __builtin_convertvector(lo, h8v);
    h8v chi = __builtin_convertvector(hi, h8v);
    return __builtin_shufflevector(clo, chi, 0, 1, 2, 3, 4, 5, 6, 7, 8, 9, 10, 11, 12, 13, 14, 15);
}
// B 32x16: lane l -> col n=l&15; K = ((l<16)?0:16) + e, contiguous 16 halves.

#if USE_ASYNC_LDS
// Signature (from hipcc diagnostic): arg0 = v4i AS1* (global), arg1 = v4i AS3*
// (LDS), then two constant ints (offset, cpol).
__device__ __forceinline__ void async_b128(const _Float16* g, _Float16* l) {
    __builtin_amdgcn_global_load_async_to_lds_b128((AS1 v4i*)g, (AS3 v4i*)l, 0, 0);
}
#endif

// ---------------------------------------------------------------------------
// Kernel 1: weight transpose + f32->f16 convert. WT[n][k] = W[k][n], 512x512.
// ---------------------------------------------------------------------------
__global__ void prep_weights(const float* __restrict__ Wq, const float* __restrict__ Wk,
                             const float* __restrict__ Wv, const float* __restrict__ Wo,
                             _Float16* __restrict__ WqT, _Float16* __restrict__ WkT,
                             _Float16* __restrict__ WvT, _Float16* __restrict__ WoT) {
    int i = blockIdx.x * blockDim.x + threadIdx.x;   // 0 .. 512*512-1
    int n = i >> 9, k = i & 511;
    WqT[i] = (_Float16)Wq[k * 512 + n];
    WkT[i] = (_Float16)Wk[k * 512 + n];
    WvT[i] = (_Float16)Wv[k * 512 + n];
    WoT[i] = (_Float16)Wo[k * 512 + n];
}

// ---------------------------------------------------------------------------
// Kernel 2: Q/K/V projections. grid.z: 0=Q, 1=K'(+R), 2=V(transposed store).
// GEMM [BT x F]*[F x HD]; wave = 16x128 strip; 8 WMMAs per K=32 step (A-frag
// reused 8x so the f32 activation stream is read/converted half as often).
// ---------------------------------------------------------------------------
__global__ __launch_bounds__(128) void qkv_proj(
    const float* __restrict__ q, const float* __restrict__ k, const float* __restrict__ v,
    const _Float16* __restrict__ WqT, const _Float16* __restrict__ WkT,
    const _Float16* __restrict__ WvT,
    const float* __restrict__ bq, const float* __restrict__ bk, const float* __restrict__ bv,
    const float* __restrict__ R,
    _Float16* __restrict__ Qs, _Float16* __restrict__ Kp, _Float16* __restrict__ Vt) {
    using namespace cfg;
    const int lane = threadIdx.x & 31;
    const int wv   = threadIdx.x >> 5;
    const int nlo  = lane & 15;
    const int up   = lane >> 4;             // half-wave select
    const int tile = blockIdx.x * 4 + wv;   // 0..2047
    const int rowbase = (tile >> 2) * 16;   // 512 row tiles
    const int nbase   = (tile & 3) * 128;   // 4 N strips of 128
    const int which   = blockIdx.z;

    const float*    X    = which == 0 ? q : which == 1 ? k : v;
    const _Float16* WT   = which == 0 ? WqT : which == 1 ? WkT : WvT;
    const float*    bias = which == 0 ? bq : which == 1 ? bk : bv;

    const float* arow = X + (size_t)(rowbase + nlo) * F;
    f8v acc[8] = {f8v{}, f8v{}, f8v{}, f8v{}, f8v{}, f8v{}, f8v{}, f8v{}};

    for (int kk = 0; kk < F; kk += 32) {
        if (kk + 64 < F) __builtin_prefetch(arow + kk + 64, 0, 1);
        h16v a = frag_a_f32(arow, kk + up * 16);
        const int kb2 = kk + up * 16;
#pragma unroll
        for (int nt = 0; nt < 8; ++nt) {
            h16v b = *(const h16v*)(WT + (size_t)(nbase + nt * 16 + nlo) * F + kb2);
            acc[nt] = WMMA_F16(a, b, acc[nt]);
        }
    }

    const int roff = up * 8;
#pragma unroll
    for (int nt = 0; nt < 8; ++nt) {
        const int hd = nbase + nt * 16 + nlo;
        const int hh = hd >> 6, dd = hd & 63;
        const float bb = bias[hd];
#pragma unroll
        for (int r = 0; r < 8; ++r) {
            const int grow = rowbase + roff + r;       // [0, B*T)
            const int bI = grow >> 11, tt = grow & (T - 1);
            const float val = acc[nt][r] + bb;
            if (which == 0) {          // Q pre-scaled by 1/sqrt(D)
                Qs[(((size_t)bI * H + hh) * T + tt) * D + dd] = (_Float16)(val * 0.125f);
            } else if (which == 1) {   // K' = K + R  (folds Q.R^T into Q.K'^T)
                Kp[(((size_t)bI * H + hh) * T + tt) * D + dd] =
                    (_Float16)(val + R[tt * D + dd]);
            } else {                   // V stored transposed [B,H,D,T]
                Vt[(((size_t)bI * H + hh) * D + dd) * T + tt] = (_Float16)val;
            }
        }
    }
}

// ---------------------------------------------------------------------------
// Kernel 3: flash attention. Block = 4 waves covering 64 query rows.
// Async path: K'/V 32-column tiles double-buffered in LDS via
// GLOBAL_LOAD_ASYNC_TO_LDS_B128 (4 ops/wave/tile, s_wait_asynccnt + barrier),
// shared by all 4 waves. Per j-step: 4 WMMAs for S, online softmax with
// shfl_xor reductions, P transposed via per-wave LDS, 4 WMMAs for O += P.V.
// ---------------------------------------------------------------------------
__global__ __launch_bounds__(128) void flash_attn(
    const _Float16* __restrict__ Qs, const _Float16* __restrict__ Kp,
    const _Float16* __restrict__ Vt, const unsigned char* __restrict__ mask,
    const float* __restrict__ qrb, _Float16* __restrict__ Oht) {
    using namespace cfg;
    __shared__ alignas(32) _Float16 Pbuf[4][16 * 32];
#if USE_ASYNC_LDS
    __shared__ alignas(64) _Float16 KT[2][32 * 64];   // [j 0..31][d 0..63]
    __shared__ alignas(64) _Float16 VT[2][64 * 32];   // [d 0..63][j 0..31]
#endif

    const int tid  = threadIdx.x;
    const int lane = tid & 31;
    const int wv   = tid >> 5;
    const int nlo  = lane & 15;
    const int up   = lane >> 4;
    const int roff = up * 8;
    const int b = blockIdx.z, h = blockIdx.y;
    const int ibase = blockIdx.x * 64 + wv * 16;

    const _Float16* Qbh = Qs + ((size_t)b * H + h) * T * D;
    const _Float16* Kbh = Kp + ((size_t)b * H + h) * T * D;
    const _Float16* Vbh = Vt + ((size_t)b * H + h) * D * T;

    // Q fragments stay in registers for the whole j-loop (D=64 -> 2 frags)
    const _Float16* qrow = Qbh + (size_t)(ibase + nlo) * D;
    const h16v qa0 = frag_a_f16(qrow, 0 + up * 16);
    const h16v qa1 = frag_a_f16(qrow, 32 + up * 16);

    f8v o[4] = {f8v{}, f8v{}, f8v{}, f8v{}};
    float mrow[8], lrow[8];
#pragma unroll
    for (int r = 0; r < 8; ++r) { mrow[r] = -3.0e38f; lrow[r] = 0.0f; }

    _Float16* myP = Pbuf[wv];

#if USE_ASYNC_LDS
    // ---- unified causal bound for the whole block (extra tiles for the
    // lower strips are fully masked; online softmax absorbs them exactly) ----
    const int jEnd = blockIdx.x * 64 + 63;

    auto issue_tile = [&](int j0, int bi) {
#pragma unroll
        for (int c0 = 0; c0 < 2; ++c0) {
            const int c = tid + c0 * 128;            // 256 B128 chunks per tile
            const int krow = c >> 3, kcol = (c & 7) * 8;
            async_b128(Kbh + (size_t)(j0 + krow) * D + kcol, &KT[bi][krow * 64 + kcol]);
            const int vd = c >> 2, vc = (c & 3) * 8;
            async_b128(Vbh + (size_t)vd * T + j0 + vc, &VT[bi][vd * 32 + vc]);
        }
    };

    issue_tile(0, 0);
    int cur = 0;
    for (int j0 = 0; j0 <= jEnd; j0 += 32) {
        const bool more = (j0 + 32) <= jEnd;
        if (more) issue_tile(j0 + 32, cur ^ 1);
        // async loads complete in order: <=4 outstanding => current tile landed
        if (more) __builtin_amdgcn_s_wait_asynccnt(4);   // imm-only argument
        else      __builtin_amdgcn_s_wait_asynccnt(0);
        __syncthreads();
        if (more) {   // prefetch next tile's bias / mask rows (HBM stream)
            const int prow = ibase + nlo;
            if (up) __builtin_prefetch(&qrb[((size_t)h * T + prow) * T + j0 + 32], 0, 1);
            else    __builtin_prefetch(&mask[(size_t)prow * T + j0 + 32], 0, 1);
        }
#else
    const int jEnd = ibase + 15;                     // per-wave causal bound
    for (int j0 = 0; j0 <= jEnd; j0 += 32) {
        if (j0 + 32 <= jEnd) {
            const int prow = ibase + nlo;
            if (up) __builtin_prefetch(&qrb[((size_t)h * T + prow) * T + j0 + 32], 0, 1);
            else    __builtin_prefetch(&mask[(size_t)prow * T + j0 + 32], 0, 1);
        }
#endif
        // ---- S = Q . K'^T, two 16x16 tiles covering columns j0..j0+31 ----
        f8v s[2];
#pragma unroll
        for (int st = 0; st < 2; ++st) {
#if USE_ASYNC_LDS
            const _Float16* kp = &KT[cur][(st * 16 + nlo) * 64 + up * 16];
#else
            const _Float16* kp = Kbh + (size_t)(j0 + st * 16 + nlo) * D + up * 16;
#endif
            h16v b0 = *(const h16v*)(kp);        // K(d) = 0..31
            h16v b1 = *(const h16v*)(kp + 32);   // K(d) = 32..63
            f8v ss = {};
            ss = WMMA_F16(qa0, b0, ss);
            ss = WMMA_F16(qa1, b1, ss);
            s[st] = ss;
        }
        // ---- add qr_bias/sqrt(D), apply boolean mask ----
#pragma unroll
        for (int st = 0; st < 2; ++st) {
            const int col = j0 + st * 16 + nlo;
#pragma unroll
            for (int r = 0; r < 8; ++r) {
                const int row = ibase + roff + r;
                const float val =
                    s[st][r] + 0.125f * qrb[((size_t)h * T + row) * T + col];
                s[st][r] = mask[(size_t)row * T + col] ? val : -1.0e9f;
            }
        }
        // ---- online softmax; rows live across the 16-lane half-waves ----
#pragma unroll
        for (int r = 0; r < 8; ++r) {
            float mx = fmaxf(s[0][r], s[1][r]);
            mx = fmaxf(mx, __shfl_xor(mx, 1, 32));
            mx = fmaxf(mx, __shfl_xor(mx, 2, 32));
            mx = fmaxf(mx, __shfl_xor(mx, 4, 32));
            mx = fmaxf(mx, __shfl_xor(mx, 8, 32));
            const float mnew  = fmaxf(mrow[r], mx);
            const float alpha = __expf(mrow[r] - mnew);
            const float p0 = __expf(s[0][r] - mnew);
            const float p1 = __expf(s[1][r] - mnew);
            s[0][r] = p0;
            s[1][r] = p1;
            float rs = p0 + p1;
            rs += __shfl_xor(rs, 1, 32);
            rs += __shfl_xor(rs, 2, 32);
            rs += __shfl_xor(rs, 4, 32);
            rs += __shfl_xor(rs, 8, 32);
            lrow[r] = lrow[r] * alpha + rs;
            mrow[r] = mnew;
#pragma unroll
            for (int nt = 0; nt < 4; ++nt) o[nt][r] = o[nt][r] * alpha;
        }
        // ---- transpose P (C-layout -> A-layout) through per-wave LDS ----
#pragma unroll
        for (int st = 0; st < 2; ++st)
#pragma unroll
            for (int r = 0; r < 8; ++r)
                myP[(roff + r) * 32 + st * 16 + nlo] = (_Float16)s[st][r];
        // same-wave DS ops are in order; compiler inserts s_wait_dscnt
        const h16v pa = frag_a_f16(myP + nlo * 32, up * 16);
        // ---- O += P . V  (B-frag from transposed V, contiguous) ----
#pragma unroll
        for (int nt = 0; nt < 4; ++nt) {
#if USE_ASYNC_LDS
            h16v vb = *(const h16v*)(&VT[cur][(nt * 16 + nlo) * 32 + up * 16]);
#else
            h16v vb = *(const h16v*)(Vbh + (size_t)(nt * 16 + nlo) * T + j0 + up * 16);
#endif
            o[nt] = WMMA_F16(pa, vb, o[nt]);
        }
#if USE_ASYNC_LDS
        __syncthreads();   // everyone done reading buf[cur] before reuse
        cur ^= 1;
#endif
    }

    // ---- normalize and store O in [B,T,H*D] f16 for the output GEMM ----
#pragma unroll
    for (int r = 0; r < 8; ++r) lrow[r] = 1.0f / lrow[r];
#pragma unroll
    for (int nt = 0; nt < 4; ++nt)
#pragma unroll
        for (int r = 0; r < 8; ++r) {
            const int row = ibase + roff + r;
            Oht[((size_t)b * T + row) * HD + h * 64 + nt * 16 + nlo] =
                (_Float16)(o[nt][r] * lrow[r]);
        }
}

// ---------------------------------------------------------------------------
// Kernel 4: output projection [BT x HD]*[HD x F] + bo -> f32 out. 16x128/wave.
// ---------------------------------------------------------------------------
__global__ __launch_bounds__(128) void out_proj(
    const _Float16* __restrict__ Oht, const _Float16* __restrict__ WoT,
    const float* __restrict__ bo, float* __restrict__ out) {
    using namespace cfg;
    const int lane = threadIdx.x & 31;
    const int wv   = threadIdx.x >> 5;
    const int nlo  = lane & 15;
    const int up   = lane >> 4;
    const int tile = blockIdx.x * 4 + wv;
    const int rowbase = (tile >> 2) * 16;
    const int nbase   = (tile & 3) * 128;

    const _Float16* arow = Oht + (size_t)(rowbase + nlo) * HD;
    f8v acc[8] = {f8v{}, f8v{}, f8v{}, f8v{}, f8v{}, f8v{}, f8v{}, f8v{}};

    for (int kk = 0; kk < HD; kk += 32) {
        if (kk + 64 < HD) __builtin_prefetch(arow + kk + 64, 0, 1);
        h16v a = frag_a_f16(arow, kk + up * 16);
        const int kb2 = kk + up * 16;
#pragma unroll
        for (int nt = 0; nt < 8; ++nt) {
            h16v b = *(const h16v*)(WoT + (size_t)(nbase + nt * 16 + nlo) * HD + kb2);
            acc[nt] = WMMA_F16(a, b, acc[nt]);
        }
    }

    const int roff = up * 8;
#pragma unroll
    for (int nt = 0; nt < 8; ++nt) {
        const int n = nbase + nt * 16 + nlo;
        const float bb = bo[n];
#pragma unroll
        for (int r = 0; r < 8; ++r)
            out[(size_t)(rowbase + roff + r) * F + n] = acc[nt][r] + bb;
    }
}

// ---------------------------------------------------------------------------
extern "C" void kernel_launch(void* const* d_in, const int* in_sizes, int n_in,
                              void* d_out, int out_size, void* d_ws, size_t ws_size,
                              hipStream_t stream) {
    using namespace cfg;
    const float* q  = (const float*)d_in[0];
    const float* k  = (const float*)d_in[1];
    const float* v  = (const float*)d_in[2];
    const unsigned char* mask = (const unsigned char*)d_in[3];   // [1,1,T,T] bool
    const float* Wq = (const float*)d_in[4];
    const float* bq = (const float*)d_in[5];
    const float* Wk = (const float*)d_in[6];
    const float* bk = (const float*)d_in[7];
    const float* Wv = (const float*)d_in[8];
    const float* bv = (const float*)d_in[9];
    const float* R  = (const float*)d_in[10];
    const float* qrb = (const float*)d_in[11];                   // [H,T,T]
    const float* Wo = (const float*)d_in[12];
    const float* bo = (const float*)d_in[13];
    float* out = (float*)d_out;

    // workspace carve-out (all offsets 256B aligned)
    char* ws = (char*)d_ws;
    size_t off = 0;
    auto take = [&](size_t bytes) { char* p = ws + off; off += (bytes + 255) & ~(size_t)255; return p; };
    _Float16* WqT = (_Float16*)take((size_t)F * HD * 2);
    _Float16* WkT = (_Float16*)take((size_t)F * HD * 2);
    _Float16* WvT = (_Float16*)take((size_t)F * HD * 2);
    _Float16* WoT = (_Float16*)take((size_t)HD * F * 2);
    _Float16* Qs  = (_Float16*)take((size_t)B * H * T * D * 2);
    _Float16* Kp  = (_Float16*)take((size_t)B * H * T * D * 2);
    _Float16* Vt  = (_Float16*)take((size_t)B * H * D * T * 2);
    _Float16* Oht = (_Float16*)take((size_t)B * T * HD * 2);

    prep_weights<<<(F * HD) / 256, 256, 0, stream>>>(Wq, Wk, Wv, Wo, WqT, WkT, WvT, WoT);

    // 512 row-tiles * 4 N-strips(128) = 2048 wave-tiles, 4 waves/block
    qkv_proj<<<dim3(512, 1, 3), 128, 0, stream>>>(q, k, v, WqT, WkT, WvT,
                                                  bq, bk, bv, R, Qs, Kp, Vt);

    flash_attn<<<dim3(T / 64, H, B), 128, 0, stream>>>(Qs, Kp, Vt, mask, qrb, Oht);

    out_proj<<<dim3(512), 128, 0, stream>>>(Oht, WoT, bo, out);
}
